// TemporalAttention_70497593197379
// MI455X (gfx1250) — compile-verified, matrix operand order checked by reading, not run on hardware
//
#include <hip/hip_runtime.h>
#include <hip/hip_bf16.h>
#include <cstdint>

#define B_   8
#define L_   273
#define DIM_ 1536
#define NH   12
#define DH   128
#define ZS_  256
#define FH   32
#define EPSV 1e-6f
#define ROWS (B_ * L_)          // 2184
#define KSTEPS (DIM_ / 32)      // 48

typedef __attribute__((ext_vector_type(16))) __bf16 v16bf;
typedef __attribute__((ext_vector_type(8)))  __bf16 v8bf;
typedef __attribute__((ext_vector_type(4)))  __bf16 v4bf;
typedef __attribute__((ext_vector_type(8)))  float  v8f;
typedef int v4i_ __attribute__((vector_size(16)));   // matches builtin's param type

#if defined(__HIP_DEVICE_COMPILE__) && __has_builtin(__builtin_amdgcn_global_load_async_to_lds_b128)
#define ASYNC_CP 1
#else
#define ASYNC_CP 0
#endif

static __device__ inline __bf16 f2bf(float f) {
  union { float f; unsigned u; } a; a.f = f;
  unsigned r = (a.u + 0x7FFFu + ((a.u >> 16) & 1u)) >> 16;
  union { unsigned short s; __bf16 b; } o; o.s = (unsigned short)r;
  return o.b;
}

// 16B global -> LDS copy: async (ASYNCcnt) when available, else via VGPRs.
static __device__ inline void lds_cp_b128(const __bf16* src, __bf16* dst) {
#if ASYNC_CP
  __builtin_amdgcn_global_load_async_to_lds_b128(
      (__attribute__((address_space(1))) v4i_*)(uintptr_t)src,
      (__attribute__((address_space(3))) v4i_*)(unsigned)(uintptr_t)dst,
      0, 0);
#else
  *reinterpret_cast<uint4*>(dst) = *reinterpret_cast<const uint4*>(src);
#endif
}

static __device__ inline void wait_async_cp() {
#if ASYNC_CP
#if __has_builtin(__builtin_amdgcn_s_wait_asynccnt)
  __builtin_amdgcn_s_wait_asynccnt(0);
#else
  asm volatile("s_wait_asynccnt 0x0" ::: "memory");
#endif
#endif
}

// ---------------------------------------------------------------------------
// f32 -> bf16 bulk convert (vectorized x4)
// ---------------------------------------------------------------------------
__global__ __launch_bounds__(256) void cvt_bf16(const float* __restrict__ s,
                                                __bf16* __restrict__ d, int n4)
{
  const int i = blockIdx.x * blockDim.x + threadIdx.x;
  if (i < n4) {
    const float4 f = reinterpret_cast<const float4*>(s)[i];
    v4bf o; o[0] = f2bf(f.x); o[1] = f2bf(f.y); o[2] = f2bf(f.z); o[3] = f2bf(f.w);
    *reinterpret_cast<v4bf*>(d + (size_t)i * 4) = o;
  }
}

// ---------------------------------------------------------------------------
// GEMM: C[M,N] = A[M,K](bf16) * W[N,K](bf16)^T + bias, 2184 x 1536 x 1536.
// 256 thr = 8 waves; tile 128x128, BK=32; wave grid 4(M) x 2(N) -> 2x4 WMMA
// tiles/wave. LDS double-buffered; tiles staged with async global->LDS b128.
// Hot loop per wave: 8 v_wmma + 12 ds_load_b128 + 4 async b128, no converts.
// ---------------------------------------------------------------------------
__global__ __launch_bounds__(256) void gemm_bf16_wmma(
    const __bf16* __restrict__ A, const __bf16* __restrict__ W,
    const float* __restrict__ bias, float* __restrict__ C)
{
  __shared__ __bf16 Ab[2][128 * 40];   // stride 40 (80B): 16B-aligned rows, no bank conflicts
  __shared__ __bf16 Bb[2][128 * 40];

  const int M = ROWS, K = DIM_;
  const int rowBase = blockIdx.y * 128;
  const int colBase = blockIdx.x * 128;
  const int t    = threadIdx.x;
  const int lane = t & 31;
  const int wave = t >> 5;
  const int wm = wave & 3;          // 32-row strip
  const int wn = wave >> 2;         // 64-col strip

  v8f acc[2][4] = {};

  // prologue: stage K-tile 0 into buffer 0
#pragma unroll
  for (int j = 0; j < 2; ++j) {
    const int c = t * 2 + j, r = c >> 2, q = (c & 3) * 8;
    if (rowBase + r < M)
      lds_cp_b128(A + (size_t)(rowBase + r) * K + q, &Ab[0][r * 40 + q]);
    lds_cp_b128(W + (size_t)(colBase + r) * K + q, &Bb[0][r * 40 + q]);
  }
  wait_async_cp();
  __syncthreads();

  for (int kt = 0; kt < KSTEPS; ++kt) {
    const int buf = kt & 1;
    // stage next K-tile into the other buffer (overlapped with WMMA below)
    if (kt + 1 < KSTEPS) {
      const int k1 = (kt + 1) * 32;
#pragma unroll
      for (int j = 0; j < 2; ++j) {
        const int c = t * 2 + j, r = c >> 2, q = (c & 3) * 8;
        if (rowBase + r < M)
          lds_cp_b128(A + (size_t)(rowBase + r) * K + k1 + q, &Ab[buf ^ 1][r * 40 + q]);
        lds_cp_b128(W + (size_t)(colBase + r) * K + k1 + q, &Bb[buf ^ 1][r * 40 + q]);
      }
    }
    if (kt + 2 < KSTEPS) {   // L2 prefetch two tiles ahead (global_prefetch_b8)
      const int r0 = (t * 2) >> 2;
      if (rowBase + r0 < M)
        __builtin_prefetch(A + (size_t)(rowBase + r0) * K + (kt + 2) * 32, 0, 0);
      __builtin_prefetch(W + (size_t)(colBase + r0) * K + (kt + 2) * 32, 0, 0);
    }

    // fragments per ISA 16-bit layout: lane<16 -> K 0..7 & 16..23 (kb=0),
    // lane>=16 -> K 8..15 & 24..31 (kb=8). B mirrored (W rows run along K).
    const int kb = (lane >> 4) * 8;
    const int rl = lane & 15;
    v16bf af[2], bfr[4];
#pragma unroll
    for (int mi = 0; mi < 2; ++mi) {
      const __bf16* p = &Ab[buf][(wm * 32 + mi * 16 + rl) * 40 + kb];
      const v8bf c0 = *reinterpret_cast<const v8bf*>(p);
      const v8bf c1 = *reinterpret_cast<const v8bf*>(p + 16);
#pragma unroll
      for (int j = 0; j < 8; ++j) { af[mi][j] = c0[j]; af[mi][8 + j] = c1[j]; }
    }
#pragma unroll
    for (int ni = 0; ni < 4; ++ni) {
      const __bf16* p = &Bb[buf][(wn * 64 + ni * 16 + rl) * 40 + kb];
      const v8bf c0 = *reinterpret_cast<const v8bf*>(p);
      const v8bf c1 = *reinterpret_cast<const v8bf*>(p + 16);
#pragma unroll
      for (int j = 0; j < 8; ++j) { bfr[ni][j] = c0[j]; bfr[ni][8 + j] = c1[j]; }
    }
#pragma unroll
    for (int mi = 0; mi < 2; ++mi)
#pragma unroll
      for (int ni = 0; ni < 4; ++ni)
        acc[mi][ni] = __builtin_amdgcn_wmma_f32_16x16x32_bf16(
            false, af[mi], false, bfr[ni], (short)0, acc[mi][ni], false, false);

    wait_async_cp();
    __syncthreads();
  }

  // epilogue: C/D layout — lane<16: M=v, N=lane ; lane>=16: M=v+8, N=lane-16
  const int rsel = (lane >> 4) * 8;
  const int csel = lane & 15;
#pragma unroll
  for (int mi = 0; mi < 2; ++mi)
#pragma unroll
    for (int ni = 0; ni < 4; ++ni) {
      const int col = colBase + wn * 64 + ni * 16 + csel;
      const float bv = bias[col];
#pragma unroll
      for (int v = 0; v < 8; ++v) {
        const int row = rowBase + wm * 32 + mi * 16 + rsel + v;
        if (row < M) C[(size_t)row * DIM_ + col] = acc[mi][ni][v] + bv;
      }
    }
}

// ---------------------------------------------------------------------------
// cos/sin tables (runtime read_rope_indices / current_frame_idx live in dev mem)
// ---------------------------------------------------------------------------
__global__ void prep_cs(const float* __restrict__ rope_table,
                        const int* __restrict__ rri, const int* __restrict__ cfi_p,
                        float2* __restrict__ cs_his, float2* __restrict__ cs_last)
{
  const int idx = blockIdx.x * blockDim.x + threadIdx.x;
  const int nh = FH * L_ * (DH / 2);
  if (idx < nh) {
    const int g = idx / (L_ * 64);
    const int rem = idx - g * (L_ * 64);
    const float f = rope_table[(size_t)rri[g] * (L_ * 64) + rem];
    cs_his[idx] = make_float2(__cosf(f), __sinf(f));
  } else if (idx - nh < L_ * 64) {
    const int i2 = idx - nh;
    const float f = rope_table[(size_t)cfi_p[0] * (L_ * 64) + i2];
    cs_last[i2] = make_float2(__cosf(f), __sinf(f));
  }
}

// ---------------------------------------------------------------------------
// RMSNorm + RoPE in-place on q,k rows; copy v_cond rows (l>=ZS) into bf16 aib.
// ---------------------------------------------------------------------------
__global__ __launch_bounds__(256) void postproc(
    float* __restrict__ q, float* __restrict__ k, const float* __restrict__ v,
    const float* __restrict__ gq, const float* __restrict__ gk,
    const float2* __restrict__ cs_his, const float2* __restrict__ cs_last,
    const int* __restrict__ cfi_p, __bf16* __restrict__ aib)
{
  __shared__ float rq[256], rk[256];
  const int row = blockIdx.x;
  const int l = row % L_;
  const int t = threadIdx.x;
  const int slot = cfi_p[0] % FH;
  float* qr = q + (size_t)row * DIM_;
  float* kr = k + (size_t)row * DIM_;

  float qv[6], kv[6], sq = 0.f, sk = 0.f;
#pragma unroll
  for (int j = 0; j < 6; ++j) {
    qv[j] = qr[t * 6 + j]; kv[j] = kr[t * 6 + j];
    sq += qv[j] * qv[j];   sk += kv[j] * kv[j];
  }
  rq[t] = sq; rk[t] = sk;
  __syncthreads();
  for (int s = 128; s > 0; s >>= 1) {
    if (t < s) { rq[t] += rq[t + s]; rk[t] += rk[t + s]; }
    __syncthreads();
  }
  const float nq = rsqrtf(rq[0] / (float)DIM_ + EPSV);
  const float nk = rsqrtf(rk[0] / (float)DIM_ + EPSV);

#pragma unroll
  for (int p = 0; p < 3; ++p) {
    const int e0 = t * 6 + 2 * p;
    const int d2 = (e0 % DH) >> 1;
    const float q0 = qv[2*p] * nq * gq[e0], q1 = qv[2*p+1] * nq * gq[e0+1];
    const float2 cl = cs_last[l * 64 + d2];
    qr[e0]     = q0 * cl.x - q1 * cl.y;
    qr[e0 + 1] = q0 * cl.y + q1 * cl.x;
    const float k0 = kv[2*p] * nk * gk[e0], k1 = kv[2*p+1] * nk * gk[e0+1];
    const float2 ch = cs_his[(slot * L_ + l) * 64 + d2];
    kr[e0]     = k0 * ch.x - k1 * ch.y;
    kr[e0 + 1] = k0 * ch.y + k1 * ch.x;
  }
  if (l >= ZS_) {
    const float* vr = v + (size_t)row * DIM_;
    __bf16* ar = aib + (size_t)row * DIM_;
#pragma unroll
    for (int j = 0; j < 6; ++j) ar[t * 6 + j] = f2bf(vr[t * 6 + j]);
  }
}

// ---------------------------------------------------------------------------
// Temporal attention: one wave32 per (b, z<ZS, n); each lane owns a float4
// D-chunk -> fully-coalesced 512B k/v frame reads, butterfly-shfl score
// reduction, per-lane redundant softmax, reduction-free output (bf16).
// ---------------------------------------------------------------------------
__global__ __launch_bounds__(256) void attn(
    const float* __restrict__ qn, const float* __restrict__ kn,
    const float* __restrict__ vn,
    const float* __restrict__ cache_k, const float* __restrict__ cache_v,
    const float2* __restrict__ cs_his, const int* __restrict__ cfi_p,
    __bf16* __restrict__ aib)
{
  const int w    = blockIdx.x * 8 + (threadIdx.x >> 5);
  const int lane = threadIdx.x & 31;
  const int n  = w % NH;
  const int zb = w / NH;
  const int z  = zb % ZS_;
  const int b  = zb / ZS_;
  const int cfi   = cfi_p[0];
  const int slot  = cfi % FH;
  const int valid = (cfi + 1 < FH) ? (cfi + 1) : FH;
  const float scale = 0.08838834764831845f;   // 1/sqrt(128)

  const size_t qoff = (size_t)(b * L_ + z) * DIM_ + n * DH + lane * 4;
  const float4 q4 = *reinterpret_cast<const float4*>(qn + qoff);

  float sc[FH];
#pragma unroll
  for (int g = 0; g < FH; ++g) {
    if (g < valid) {
      float4 k4;
      if (g == slot) {
        k4 = *reinterpret_cast<const float4*>(kn + qoff);   // already roped
      } else {
        const size_t ko = ((size_t)((b * FH + g) * L_ + z) * NH + n) * DH + lane * 4;
        k4 = *reinterpret_cast<const float4*>(cache_k + ko);
        const float2 c0 = cs_his[((g * L_ + z) << 6) + lane * 2];
        const float2 c1 = cs_his[((g * L_ + z) << 6) + lane * 2 + 1];
        const float x0 = k4.x, x1 = k4.y, x2 = k4.z, x3 = k4.w;
        k4.x = x0 * c0.x - x1 * c0.y;  k4.y = x0 * c0.y + x1 * c0.x;
        k4.z = x2 * c1.x - x3 * c1.y;  k4.w = x2 * c1.y + x3 * c1.x;
      }
      float partial = k4.x * q4.x + k4.y * q4.y + k4.z * q4.z + k4.w * q4.w;
#pragma unroll
      for (int off = 16; off > 0; off >>= 1) partial += __shfl_xor(partial, off, 32);
      sc[g] = partial * scale;
    } else {
      sc[g] = -1e30f;
    }
  }
  float m = -1e30f;
#pragma unroll
  for (int g = 0; g < FH; ++g) m = fmaxf(m, sc[g]);
  float ssum = 0.f;
#pragma unroll
  for (int g = 0; g < FH; ++g) { const float e = __expf(sc[g] - m); sc[g] = e; ssum += e; }
  const float inv = 1.f / ssum;

  float4 acc = make_float4(0.f, 0.f, 0.f, 0.f);
#pragma unroll
  for (int g = 0; g < FH; ++g) {
    if (g < valid) {
      float4 v4;
      if (g == slot) v4 = *reinterpret_cast<const float4*>(vn + qoff);
      else {
        const size_t vo = ((size_t)((b * FH + g) * L_ + z) * NH + n) * DH + lane * 4;
        v4 = *reinterpret_cast<const float4*>(cache_v + vo);
      }
      const float pg = sc[g] * inv;
      acc.x += pg * v4.x; acc.y += pg * v4.y; acc.z += pg * v4.z; acc.w += pg * v4.w;
    }
  }
  v4bf o; o[0] = f2bf(acc.x); o[1] = f2bf(acc.y); o[2] = f2bf(acc.z); o[3] = f2bf(acc.w);
  *reinterpret_cast<v4bf*>(aib + qoff) = o;
}

// ---------------------------------------------------------------------------
extern "C" void kernel_launch(void* const* d_in, const int* in_sizes, int n_in,
                              void* d_out, int out_size, void* d_ws, size_t ws_size,
                              hipStream_t stream)
{
  const float* x       = (const float*)d_in[0];
  const float* w_q     = (const float*)d_in[1];
  const float* b_q     = (const float*)d_in[2];
  const float* w_k     = (const float*)d_in[3];
  const float* b_k     = (const float*)d_in[4];
  const float* w_v     = (const float*)d_in[5];
  const float* b_v     = (const float*)d_in[6];
  const float* w_o     = (const float*)d_in[7];
  const float* b_o     = (const float*)d_in[8];
  const float* g_q     = (const float*)d_in[9];
  const float* g_k     = (const float*)d_in[10];
  const float* cache_k = (const float*)d_in[11];
  const float* cache_v = (const float*)d_in[12];
  const float* rope_t  = (const float*)d_in[13];
  const int*   rri     = (const int*)d_in[14];
  const int*   cfi     = (const int*)d_in[15];
  float* out = (float*)d_out;

  char* ws = (char*)d_ws;
  const size_t RM  = (size_t)ROWS * DIM_;        // activation elements
  const size_t WM  = (size_t)DIM_ * DIM_;        // weight elements
  float*  ws_q  = (float*)ws;  ws += RM * 4;
  float*  ws_k  = (float*)ws;  ws += RM * 4;
  float*  ws_v  = (float*)ws;  ws += RM * 4;
  __bf16* xb    = (__bf16*)ws; ws += RM * 2;
  __bf16* aib   = (__bf16*)ws; ws += RM * 2;
  __bf16* wqb   = (__bf16*)ws; ws += WM * 2;
  __bf16* wkb   = (__bf16*)ws; ws += WM * 2;
  __bf16* wvb   = (__bf16*)ws; ws += WM * 2;
  __bf16* wob   = (__bf16*)ws; ws += WM * 2;
  float2* cs_his  = (float2*)ws; ws += (size_t)FH * L_ * 64 * sizeof(float2);
  float2* cs_last = (float2*)ws;

  // one-time bf16 conversion of GEMM operands
  const int xr4 = (int)(RM / 4), wr4 = (int)(WM / 4);
  cvt_bf16<<<(xr4 + 255) / 256, 256, 0, stream>>>(x,   xb,  xr4);
  cvt_bf16<<<(wr4 + 255) / 256, 256, 0, stream>>>(w_q, wqb, wr4);
  cvt_bf16<<<(wr4 + 255) / 256, 256, 0, stream>>>(w_k, wkb, wr4);
  cvt_bf16<<<(wr4 + 255) / 256, 256, 0, stream>>>(w_v, wvb, wr4);
  cvt_bf16<<<(wr4 + 255) / 256, 256, 0, stream>>>(w_o, wob, wr4);

  const int cs_elems = FH * L_ * 64 + L_ * 64;
  prep_cs<<<(cs_elems + 255) / 256, 256, 0, stream>>>(rope_t, rri, cfi, cs_his, cs_last);

  dim3 gg(DIM_ / 128, (ROWS + 127) / 128);   // 12 x 18
  gemm_bf16_wmma<<<gg, 256, 0, stream>>>(xb, wqb, b_q, ws_q);
  gemm_bf16_wmma<<<gg, 256, 0, stream>>>(xb, wkb, b_k, ws_k);
  gemm_bf16_wmma<<<gg, 256, 0, stream>>>(xb, wvb, b_v, ws_v);

  postproc<<<ROWS, 256, 0, stream>>>(ws_q, ws_k, ws_v, g_q, g_k, cs_his, cs_last, cfi, aib);

  attn<<<(B_ * ZS_ * NH) / 8, 256, 0, stream>>>(ws_q, ws_k, ws_v, cache_k, cache_v,
                                                cs_his, cfi, aib);

  gemm_bf16_wmma<<<gg, 256, 0, stream>>>(aib, wob, b_o, out);
}
